// Stream_PreNet_9242769622415
// MI455X (gfx1250) — matrix operand order repaired and checked by reading, not compiled, not run on hardware
//
#include <hip/hip_runtime.h>

// ---------------------------------------------------------------------------
// Fused Stream_PreNet for gfx1250:
//   conv [1,7] (8->96, symmetric pad in W)  then  conv [7,1] (96->96, sym pad in H)
// One workgroup computes a 16x16 pixel tile x 96 channels, keeping the conv1
// intermediate (22 rows x 16 cols x 96ch, f16) entirely in LDS -> single pass
// over HBM (~436 MB total => ~19us at 23.3 TB/s).  Matrix math uses
// v_wmma_f32_16x16x32_f16 (f32 accumulate); weights are staged into LDS with
// gfx1250 async copies (global_load_async_to_lds_b128, ASYNCcnt).
// ---------------------------------------------------------------------------

typedef __attribute__((ext_vector_type(16))) _Float16 v16h;
typedef __attribute__((ext_vector_type(8)))  float    v8f;

union FragU { v16h v; uint4 q[2]; };

__device__ __forceinline__ unsigned short f2h(float f) {
    union { _Float16 h; unsigned short u; } t;
    t.h = (_Float16)f;
    return t.u;
}

__device__ __forceinline__ int symidx(int i, int n) {
    i = (i < 0) ? (-i - 1) : i;
    i = (i >= n) ? (2 * n - 1 - i) : i;
    return i;
}

// One per-lane async 16B copy global -> LDS (tracked by ASYNCcnt).
__device__ __forceinline__ void async_b128_to_lds(unsigned lds_byte_off,
                                                  const void* gptr) {
    asm volatile("global_load_async_to_lds_b128 %0, %1, off"
                 :
                 : "v"(lds_byte_off), "v"(gptr)
                 : "memory");
}

__device__ __forceinline__ void wait_asynccnt0() {
    asm volatile("s_wait_asynccnt 0" ::: "memory");
}

// ---------------------------------------------------------------------------
// Pre-pass: convert W1 [1,7,8,96] and W2 [7,1,96,96] (f32, HWIO) into f16
// B-matrix fragments for V_WMMA_F32_16X16X32_F16, in this code's contiguous k order (A matches):
//   lane l (<16): N = l,  K = khalf*16 + e  (khalf = lane>>4, e = 0..15)
// Fragment array layout: [chunk][ntile][lane][e] (16 f16 = 32 B per lane).
// W1frag: 2 chunks (K=56 padded to 64), W2frag: 21 chunks (K=672).
// ---------------------------------------------------------------------------
__global__ void prenet_prep_weights(const float* __restrict__ W1,
                                    const float* __restrict__ W2,
                                    unsigned short* __restrict__ wfrag) {
    int i = blockIdx.x * 256 + threadIdx.x;
    if (i < 6144) {                      // W1 frags: 2*6*32*16
        int e    = i & 15;
        int lane = (i >> 4) & 31;
        int cn   = i >> 9;               // chunk*6 + ntile
        int nt   = cn % 6, ch = cn / 6;
        int n    = nt * 16 + (lane & 15);
        int k    = ch * 32 + (lane >> 4) * 16 + e;
        float v  = (k < 56) ? W1[k * 96 + n] : 0.0f;   // k = kw*8 + ci
        wfrag[i] = f2h(v);
    } else if (i < 70656) {              // W2 frags: 21*6*32*16
        int j    = i - 6144;
        int e    = j & 15;
        int lane = (j >> 4) & 31;
        int cn   = j >> 9;
        int nt   = cn % 6, ch = cn / 6;
        int n    = nt * 16 + (lane & 15);
        int k    = ch * 32 + (lane >> 4) * 16 + e;     // k = kh*96 + ci, k<672
        wfrag[i] = f2h(W2[k * 96 + n]);
    }
}

// LDS layout (bytes)
#define OFF_W1  0          // 2*6*32*32   = 12288
#define OFF_W2  12288      // 21*6*32*32  = 129024
#define OFF_X   141312     // 22*22*8*2   = 7744
#define OFF_MID 149056     // 22*16*96*2  = 67584
#define LDS_BYTES 216640

__global__ __launch_bounds__(256, 1)
void prenet_fused(const float* __restrict__ x,      // [8,512,256,8]
                  const float* __restrict__ b1,     // [96]
                  const float* __restrict__ b2,     // [96]
                  const unsigned short* __restrict__ wfrag,
                  float* __restrict__ out) {        // [8,512,256,96]
    extern __shared__ char smem[];

    const int tid  = threadIdx.x;
    const int lane = tid & 31;
    const int wave = tid >> 5;
    const int l    = lane & 15;        // N (or M) index within fragment
    const int kh2  = lane >> 4;        // khalf

    const int w0 = blockIdx.x * 16;
    const int h0 = blockIdx.y * 16;
    const int b  = blockIdx.z;

    // ---- stage pre-swizzled weight fragments into LDS via async copies ----
    // 141312 B = 8832 x b128; overlaps with the x-tile gather below.
    {
        const char* src = (const char*)wfrag;
        for (int i = tid; i < 8832; i += 256) {
            async_b128_to_lds((unsigned)(OFF_W1 + i * 16), src + i * 16);
        }
    }

    // ---- load 22x22x8 input patch, f32 -> f16, symmetric-pad indexing ----
    for (int p = tid; p < 484; p += 256) {
        int r  = p / 22, c = p % 22;
        int gh = symidx(h0 + r - 3, 512);
        int gw = symidx(w0 + c - 3, 256);
        const float* src = x + (((long)b * 512 + gh) * 256 + gw) * 8;
        float4 f0 = ((const float4*)src)[0];
        float4 f1 = ((const float4*)src)[1];
        union { unsigned short u[8]; uint4 q; } pk;
        pk.u[0] = f2h(f0.x); pk.u[1] = f2h(f0.y);
        pk.u[2] = f2h(f0.z); pk.u[3] = f2h(f0.w);
        pk.u[4] = f2h(f1.x); pk.u[5] = f2h(f1.y);
        pk.u[6] = f2h(f1.z); pk.u[7] = f2h(f1.w);
        *(uint4*)(smem + OFF_X + p * 16) = pk.q;
    }
    wait_asynccnt0();          // weight frags resident in LDS
    __syncthreads();

    // ---- conv1 [1,7] 8->96 : 22 mid rows x 6 N-tiles, K = 56 (2 chunks) ----
    // unit u = r*6 + nt; stride 8 == (+1 row, +2 ntiles), incremental decomp.
    {
        int r = wave / 6, nt = wave % 6;
        for (int u = wave; u < 132; u += 8) {
            float bv = b1[nt * 16 + l];
            v8f acc;
#pragma unroll
            for (int i = 0; i < 8; ++i) acc[i] = bv;

#pragma unroll
            for (int ch = 0; ch < 2; ++ch) {
                FragU a, bf;
#pragma unroll
                for (int g = 0; g < 2; ++g) {
                    // A 16x32 f16: k = khalf*16 + g*8 + (0..7); a K-group of 8
                    // contiguous ci is one 16B LDS chunk at xtile[r][m+kw][0..7]
                    int kbase = ch * 32 + kh2 * 16 + g * 8;
                    int kw    = kbase >> 3;                 // k = kw*8 + ci
                    bool pad  = (kw >= 7);                  // K padding 56..63
                    int col   = l + (pad ? 0 : kw);
                    uint4 qa  = *(const uint4*)(smem + OFF_X + (r * 22 + col) * 16);
                    if (pad) { qa.x = 0u; qa.y = 0u; qa.z = 0u; qa.w = 0u; }
                    a.q[g] = qa;
                }
                const uint4* pb =
                    (const uint4*)(smem + OFF_W1 + ((ch * 6 + nt) * 32 + lane) * 32);
                bf.q[0] = pb[0];
                bf.q[1] = pb[1];
                acc = __builtin_amdgcn_wmma_f32_16x16x32_f16(
                    false, a.v, false, bf.v, (short)0, acc, false, false);
            }
            // D layout: VGPR i -> M = i + 8*khalf (w index), N = l (channel)
#pragma unroll
            for (int i = 0; i < 8; ++i) {
                int m = i + kh2 * 8;
                *(unsigned short*)(smem + OFF_MID +
                                   ((r * 16 + m) * 96 + nt * 16 + l) * 2) =
                    f2h(acc[i]);
            }
            nt += 2; r += 1;
            if (nt >= 6) { nt -= 6; r += 1; }
        }
    }
    __syncthreads();

    // ---- conv2 [7,1] 96->96 : 16 out rows x 6 N-tiles, K = 672 (21 chunks) --
    {
        int r = wave / 6, nt = wave % 6;
        for (int u = wave; u < 96; u += 8) {
            float bv = b2[nt * 16 + l];
            v8f acc;
#pragma unroll
            for (int i = 0; i < 8; ++i) acc[i] = bv;

#pragma unroll
            for (int ch = 0; ch < 21; ++ch) {
                FragU a, bf;
#pragma unroll
                for (int g = 0; g < 2; ++g) {
                    int kbase = ch * 32 + kh2 * 16 + g * 8;   // k = kh*96 + ci
                    int kh    = kbase / 96;                   // const after unroll
                    int ci    = kbase % 96;                   // multiple of 8
                    a.q[g] = *(const uint4*)(smem + OFF_MID +
                                             (((r + kh) * 16 + l) * 96 + ci) * 2);
                }
                const uint4* pb =
                    (const uint4*)(smem + OFF_W2 + ((ch * 6 + nt) * 32 + lane) * 32);
                bf.q[0] = pb[0];
                bf.q[1] = pb[1];
                acc = __builtin_amdgcn_wmma_f32_16x16x32_f16(
                    false, a.v, false, bf.v, (short)0, acc, false, false);
            }
            float* op =
                out + (((long)b * 512 + (h0 + r)) * 256 + w0) * 96 + nt * 16 + l;
#pragma unroll
            for (int i = 0; i < 8; ++i) {
                op[(i + kh2 * 8) * 96] = acc[i];
            }
            nt += 2; r += 1;
            if (nt >= 6) { nt -= 6; r += 1; }
        }
    }
}

extern "C" void kernel_launch(void* const* d_in, const int* in_sizes, int n_in,
                              void* d_out, int out_size, void* d_ws, size_t ws_size,
                              hipStream_t stream) {
    const float* x  = (const float*)d_in[0];   // feature_in [8,512,256,8]
    const float* W1 = (const float*)d_in[1];   // [1,7,8,96]
    const float* b1 = (const float*)d_in[2];   // [96]
    const float* W2 = (const float*)d_in[3];   // [7,1,96,96]
    const float* b2 = (const float*)d_in[4];   // [96]
    float* out = (float*)d_out;
    unsigned short* wfrag = (unsigned short*)d_ws;   // 70656 u16 = 138 KB

    prenet_prep_weights<<<276, 256, 0, stream>>>(W1, W2, wfrag);

    dim3 grid(256 / 16, 512 / 16, 8);   // (16, 32, 8) tiles
    prenet_fused<<<grid, dim3(256), LDS_BYTES, stream>>>(x, b1, b2, wfrag, out);
}
